// GatedAttentionLayer_23184233464274
// MI455X (gfx1250) — matrix operand order repaired and checked by reading, not compiled, not run on hardware
//
#include <hip/hip_runtime.h>

// ---------------------------------------------------------------------------
// Gated attention layer for MI455X (gfx1250, wave32, WMMA).
// All heavy math goes through v_wmma_f32_16x16x32_f16 (f16 in, f32 acc).
// Pipeline: prep(add+transpose) -> WMMA GEMM1 (QKV, scatter epilogue)
//           -> flash attention (WMMA QK^T + PV, online softmax)
//           -> WMMA GEMM2 -> gating epilogue.
// GEMM is LDS double-buffered; attention software-pipelines the K-tile
// fragment loads across s-iterations and prefetches the next V tile.
// ---------------------------------------------------------------------------

typedef _Float16 half_t;
typedef __attribute__((ext_vector_type(16))) _Float16 v16h;
typedef __attribute__((ext_vector_type(8)))  float    v8f;

#define BATCH 16
#define C_DIM 512
#define T_DIM 1024
#define HEADS 8
#define HDIM  64

__device__ __forceinline__ v8f wmma16x16x32(v16h a, v16h b, v8f c) {
  return __builtin_amdgcn_wmma_f32_16x16x32_f16(false, a, false, b,
                                                (short)0, c, false, false);
}

// A fragment (16x32 f16, M x K). Lane L: row m = L&15, half h = L>>4.
// elems 0..7  -> K = 8*h + j ; elems 8..15 -> K = 16 + 8*h + j.
__device__ __forceinline__ v16h load_a_frag(const half_t* __restrict__ base,
                                            int ld, int lane) {
  const half_t* p = base + (size_t)(lane & 15) * ld + 8 * (lane >> 4);
  v16h a;
#pragma unroll
  for (int j = 0; j < 8; ++j) a[j] = p[j];
#pragma unroll
  for (int j = 0; j < 8; ++j) a[8 + j] = p[16 + j];
  return a;
}

// B fragment (32x16 f16, K x N). Lane L: col n = L&15, half h = L>>4.
// elems 0..7  -> K = 8*h + j ; elems 8..15 -> K = 16 + 8*h + j.
__device__ __forceinline__ v16h load_b_frag(const half_t* __restrict__ base,
                                            int ld, int lane) {
  const half_t* p = base + (lane & 15) + (size_t)(8 * (lane >> 4)) * ld;
  v16h b;
#pragma unroll
  for (int j = 0; j < 8; ++j) b[j] = p[(size_t)j * ld];
#pragma unroll
  for (int j = 0; j < 8; ++j) b[8 + j] = p[(size_t)(16 + j) * ld];
  return b;
}

// ---------------------------------------------------------------------------
// prep: xt[(b*T + t), c] = f16( x[b,c,t] + pos_emb[c,t] ), 32x32 LDS transpose
// ---------------------------------------------------------------------------
__global__ __launch_bounds__(256)
void ga_prep_kernel(const float* __restrict__ x, const float* __restrict__ pe,
                    half_t* __restrict__ xt) {
  __shared__ half_t tile[32][34];
  int b  = blockIdx.z;
  int c0 = blockIdx.y * 32;
  int t0 = blockIdx.x * 32;
  int tc = threadIdx.x & 31;   // fast index
  int rr = threadIdx.x >> 5;   // 8 rows / pass
#pragma unroll
  for (int i = 0; i < 4; ++i) {
    int c = c0 + rr + i * 8;
    float v = x[((size_t)b * C_DIM + c) * T_DIM + t0 + tc] +
              pe[(size_t)c * T_DIM + t0 + tc];
    tile[rr + i * 8][tc] = (half_t)v;
  }
  __syncthreads();
#pragma unroll
  for (int i = 0; i < 4; ++i) {
    int t = t0 + rr + i * 8;
    xt[((size_t)b * T_DIM + t) * C_DIM + c0 + tc] = tile[tc][rr + i * 8];
  }
}

__global__ __launch_bounds__(256)
void ga_cvt_kernel(const float* __restrict__ src, half_t* __restrict__ dst, int n) {
  int i = blockIdx.x * 256 + threadIdx.x;
  if (i < n) dst[i] = (half_t)src[i];
}

// ---------------------------------------------------------------------------
// Tiled WMMA GEMM: C[M,N] = A[M,K] * B[K,N] + bias.  LDS double-buffered.
// Block tile 128x64, BK=32, 256 threads = 8 waves in 4(M) x 2(N) grid,
// each wave computes a 32x32 tile as 2x2 WMMA fragments.
// MODE 0: scatter into Q (scaled 1/8) / Kt / V layouts.  MODE 1: plain f16.
// ---------------------------------------------------------------------------
template <int MODE>
__global__ __launch_bounds__(256)
void ga_gemm_kernel(const half_t* __restrict__ A, const half_t* __restrict__ B,
                    const float* __restrict__ bias,
                    half_t* __restrict__ outQ, half_t* __restrict__ outKt,
                    half_t* __restrict__ outV, half_t* __restrict__ outP,
                    int M, int N, int K) {
  __shared__ half_t sA[2][128][40];  // 128 x 32 (+pad), double buffered
  __shared__ half_t sB[2][32][72];   // 32 x 64 (+pad),  double buffered

  const int tid  = threadIdx.x;
  const int wave = tid >> 5;
  const int lane = tid & 31;
  const int wm = wave >> 1, wn = wave & 1;
  const int m0 = blockIdx.y * 128;
  const int n0 = blockIdx.x * 64;

  const int arow = tid >> 1, akofs = (tid & 1) * 16;  // A: 16 halves/thread
  const int brow = tid >> 3, bnofs = (tid & 7) * 8;   // B: 8 halves/thread

  const int NK = K >> 5;
  v8f acc[2][2] = {};

  // prologue: stage tile 0
  {
    const uint4* ga =
        reinterpret_cast<const uint4*>(A + (size_t)(m0 + arow) * K + akofs);
    uint4 a0 = ga[0], a1 = ga[1];
    const uint4* gb =
        reinterpret_cast<const uint4*>(B + (size_t)brow * N + n0 + bnofs);
    uint4 b0 = gb[0];
    *reinterpret_cast<uint4*>(&sA[0][arow][akofs])     = a0;
    *reinterpret_cast<uint4*>(&sA[0][arow][akofs + 8]) = a1;
    *reinterpret_cast<uint4*>(&sB[0][brow][bnofs])     = b0;
  }
  __syncthreads();

  for (int kt = 0; kt < NK; ++kt) {
    const int cur = kt & 1;
    const bool more = (kt + 1 < NK);
    uint4 a0{}, a1{}, b0{};
    if (more) {  // fetch next tile into registers while computing this one
      const int k0 = (kt + 1) << 5;
      const uint4* ga = reinterpret_cast<const uint4*>(
          A + (size_t)(m0 + arow) * K + k0 + akofs);
      a0 = ga[0];
      a1 = ga[1];
      const uint4* gb = reinterpret_cast<const uint4*>(
          B + (size_t)(k0 + brow) * N + n0 + bnofs);
      b0 = gb[0];
    }

    v16h af[2], bf[2];
    af[0] = load_a_frag(&sA[cur][wm * 32 + 0][0], 40, lane);
    af[1] = load_a_frag(&sA[cur][wm * 32 + 16][0], 40, lane);
    bf[0] = load_b_frag(&sB[cur][0][wn * 32 + 0], 72, lane);
    bf[1] = load_b_frag(&sB[cur][0][wn * 32 + 16], 72, lane);
#pragma unroll
    for (int mi = 0; mi < 2; ++mi)
#pragma unroll
      for (int ni = 0; ni < 2; ++ni)
        acc[mi][ni] = wmma16x16x32(af[mi], bf[ni], acc[mi][ni]);

    if (more) {
      *reinterpret_cast<uint4*>(&sA[cur ^ 1][arow][akofs])     = a0;
      *reinterpret_cast<uint4*>(&sA[cur ^ 1][arow][akofs + 8]) = a1;
      *reinterpret_cast<uint4*>(&sB[cur ^ 1][brow][bnofs])     = b0;
    }
    __syncthreads();
  }

  // Epilogue. D layout: lane L, vgpr r -> m = r + 8*(L>>4), n = L&15.
  const int h = lane >> 4, nl = lane & 15;
#pragma unroll
  for (int mi = 0; mi < 2; ++mi) {
#pragma unroll
    for (int ni = 0; ni < 2; ++ni) {
      const int ng = n0 + wn * 32 + ni * 16 + nl;
      const float bv = bias[ng];
#pragma unroll
      for (int r = 0; r < 8; ++r) {
        const int mg = m0 + wm * 32 + mi * 16 + 8 * h + r;
        float v = acc[mi][ni][r] + bv;
        if (MODE == 0) {
          // n = qkv*512 + head*64 + d ; m = b*1024 + t
          const int q  = ng >> 9;
          const int hh = (ng & 511) >> 6;
          const int d  = ng & 63;
          const int bb = mg >> 10;
          const int t  = mg & 1023;
          const size_t bhi = (size_t)bb * HEADS + hh;
          if (q == 0)
            outQ[(bhi * T_DIM + t) * HDIM + d] = (half_t)(v * 0.125f);
          else if (q == 1)
            outKt[(bhi * HDIM + d) * T_DIM + t] = (half_t)v;
          else
            outV[(bhi * T_DIM + t) * HDIM + d] = (half_t)v;
        } else {
          outP[(size_t)mg * N + ng] = (half_t)v;
        }
      }
    }
  }
}

// ---------------------------------------------------------------------------
// Flash attention. grid = (T/64, b*heads), 128 threads = 4 waves.
// Wave w owns 16 query rows; per 64-key step: S = Q*Kt (8 wmma), online
// softmax via wave-private LDS tile, O = O*corr + P*V (8 wmma).
// K-tile fragments are software-pipelined across s-iterations (loads for
// tile i+1 issue under tile i's PV WMMAs); V tile is prefetched
// into near cache.  Output f16 as [(b*T + t), h*64 + d].
// ---------------------------------------------------------------------------
__global__ __launch_bounds__(128)
void ga_attn_kernel(const half_t* __restrict__ Q, const half_t* __restrict__ Kt,
                    const half_t* __restrict__ V, half_t* __restrict__ O) {
  __shared__ float  sS[4][16][68];
  __shared__ half_t sP[4][16][72];
  __shared__ float  sCorr[4][16];

  const int bh   = blockIdx.y;
  const int wave = threadIdx.x >> 5;
  const int lane = threadIdx.x & 31;
  const int q0   = blockIdx.x * 64 + wave * 16;
  const int bb = bh >> 3, hh = bh & 7;

  const half_t* Qb  = Q  + (size_t)bh * T_DIM * HDIM;
  const half_t* Ktb = Kt + (size_t)bh * HDIM * T_DIM;
  const half_t* Vb  = V  + (size_t)bh * T_DIM * HDIM;

  // Q fragments (rows q0..q0+15, K split 0..31 / 32..63)
  v16h aq[2];
  aq[0] = load_a_frag(Qb + (size_t)q0 * HDIM + 0, HDIM, lane);
  aq[1] = load_a_frag(Qb + (size_t)q0 * HDIM + 32, HDIM, lane);

  v8f oacc[4] = {};
  const int row = lane & 15, chalf = lane >> 4;
  const int h = chalf, nl = lane & 15;
  float row_max = -1e30f, row_sum = 0.f;

  // K fragments for the current s-tile (pipelined across iterations)
  v16h bk0[4], bk1[4];
  auto load_bk = [&](int s) {
#pragma unroll
    for (int ni = 0; ni < 4; ++ni) {
      bk0[ni] = load_b_frag(Ktb + (size_t)0 * T_DIM + s + ni * 16, T_DIM, lane);
      bk1[ni] = load_b_frag(Ktb + (size_t)32 * T_DIM + s + ni * 16, T_DIM, lane);
    }
  };
  load_bk(0);

  for (int s0 = 0; s0 < T_DIM; s0 += 64) {
    const bool more = (s0 + 64 < T_DIM);

    // ---- scores S = Q * K^T (Q pre-scaled by 1/sqrt(d)) ----
    v8f sf[4];
#pragma unroll
    for (int ni = 0; ni < 4; ++ni) {
      v8f z = {};
      z = wmma16x16x32(aq[0], bk0[ni], z);
      sf[ni] = wmma16x16x32(aq[1], bk1[ni], z);
    }
#pragma unroll
    for (int ni = 0; ni < 4; ++ni)
#pragma unroll
      for (int r = 0; r < 8; ++r) sS[wave][r + 8 * h][ni * 16 + nl] = sf[ni][r];
    __syncthreads();

    // prefetch next V tile into near caches; fetch this tile's V fragments
    if (more)
      __builtin_prefetch(Vb + (size_t)(s0 + 64 + (threadIdx.x & 63)) * HDIM, 0, 3);
    v16h bv0[4], bv1[4];
#pragma unroll
    for (int ni = 0; ni < 4; ++ni) {
      bv0[ni] = load_b_frag(Vb + (size_t)(s0 + 0) * HDIM + ni * 16, HDIM, lane);
      bv1[ni] = load_b_frag(Vb + (size_t)(s0 + 32) * HDIM + ni * 16, HDIM, lane);
    }

    // ---- online softmax (2 lanes per row; each scans 32 of 64 cols) ----
    float tmax = -1e30f;
#pragma unroll
    for (int j = 0; j < 32; ++j)
      tmax = fmaxf(tmax, sS[wave][row][chalf * 32 + j]);
    tmax = fmaxf(tmax, __shfl_xor(tmax, 16, 32));
    const float nmax = fmaxf(row_max, tmax);
    const float corr = __expf(row_max - nmax);
    float psum = 0.f;
#pragma unroll
    for (int j = 0; j < 32; ++j) {
      float e = __expf(sS[wave][row][chalf * 32 + j] - nmax);
      psum += e;
      sP[wave][row][chalf * 32 + j] = (half_t)e;
    }
    psum += __shfl_xor(psum, 16, 32);
    row_sum = row_sum * corr + psum;
    row_max = nmax;
    if (chalf == 0) sCorr[wave][row] = corr;
    __syncthreads();

    // ---- pipeline: start next K-tile loads under the PV phase ----
    if (more) load_bk(s0 + 64);

    // ---- rescale accumulators, then O += P * V ----
    float cr[8];
#pragma unroll
    for (int r = 0; r < 8; ++r) cr[r] = sCorr[wave][r + 8 * h];
#pragma unroll
    for (int ni = 0; ni < 4; ++ni)
#pragma unroll
      for (int r = 0; r < 8; ++r) oacc[ni][r] *= cr[r];

    v16h ap0 = load_a_frag(&sP[wave][0][0], 72, lane);
    v16h ap1 = load_a_frag(&sP[wave][0][32], 72, lane);
#pragma unroll
    for (int ni = 0; ni < 4; ++ni) {
      oacc[ni] = wmma16x16x32(ap0, bv0[ni], oacc[ni]);
      oacc[ni] = wmma16x16x32(ap1, bv1[ni], oacc[ni]);
    }
    __syncthreads();
  }

  // ---- normalize and store O[(b,t), hh*64+d] ----
  if (chalf == 0) sCorr[wave][row] = 1.f / row_sum;
  __syncthreads();
  float linv[8];
#pragma unroll
  for (int r = 0; r < 8; ++r) linv[r] = sCorr[wave][r + 8 * h];
#pragma unroll
  for (int ni = 0; ni < 4; ++ni) {
#pragma unroll
    for (int r = 0; r < 8; ++r) {
      const int t = q0 + r + 8 * h;
      const int d = ni * 16 + nl;
      O[((size_t)bb * T_DIM + t) * C_DIM + hh * HDIM + d] =
          (half_t)(oacc[ni][r] * linv[r]);
    }
  }
}

// ---------------------------------------------------------------------------
// final: out[b,c,t] = x[b,c,t] + a * sigmoid(g),  a = g[(b,t),c], g = [..,512+c]
// ---------------------------------------------------------------------------
__global__ __launch_bounds__(256)
void ga_final_kernel(const float* __restrict__ x, const half_t* __restrict__ g,
                     float* __restrict__ out) {
  const size_t i = (size_t)blockIdx.x * 256 + threadIdx.x;
  const int t = (int)(i & 1023);
  const int c = (int)((i >> 10) & 511);
  const int b = (int)(i >> 19);
  const size_t grow = ((size_t)b * T_DIM + t) * (2 * C_DIM);
  const float a  = (float)g[grow + c];
  const float gp = (float)g[grow + C_DIM + c];
  const float sig = 1.f / (1.f + __expf(-gp));
  out[i] = x[i] + a * sig;
}

// ---------------------------------------------------------------------------
extern "C" void kernel_launch(void* const* d_in, const int* in_sizes, int n_in,
                              void* d_out, int out_size, void* d_ws, size_t ws_size,
                              hipStream_t stream) {
  const float* x  = (const float*)d_in[0];
  const float* pe = (const float*)d_in[1];
  const float* W1 = (const float*)d_in[2];
  const float* b1 = (const float*)d_in[3];
  const float* W2 = (const float*)d_in[4];
  const float* b2 = (const float*)d_in[5];
  float* out = (float*)d_out;
  char* ws = (char*)d_ws;

  const size_t SZ = (size_t)BATCH * T_DIM * C_DIM * sizeof(half_t);  // 16 MiB
  half_t* xt  = (half_t*)(ws + 0);       // [b*T, C]; reused as attention output
  half_t* Qb  = (half_t*)(ws + SZ);      // [bh, T, 64]
  half_t* Ktb = (half_t*)(ws + 2 * SZ);  // [bh, 64, T]
  half_t* Vb  = (half_t*)(ws + 3 * SZ);  // [bh, T, 64]
  half_t* g   = (half_t*)(ws + SZ);      // [b*T, 1024]; reuses dead Q+Kt space
  half_t* W1h = (half_t*)(ws + 4 * SZ);
  half_t* W2h = (half_t*)(ws + 4 * SZ + (size_t)C_DIM * 3 * C_DIM * sizeof(half_t));

  const int M = BATCH * T_DIM;  // 16384

  // weights -> f16
  ga_cvt_kernel<<<(C_DIM * 3 * C_DIM + 255) / 256, 256, 0, stream>>>(W1, W1h,
                                                                     C_DIM * 3 * C_DIM);
  ga_cvt_kernel<<<(C_DIM * 2 * C_DIM + 255) / 256, 256, 0, stream>>>(W2, W2h,
                                                                     C_DIM * 2 * C_DIM);
  // x + pos_emb, transpose to [b*T, C] f16
  ga_prep_kernel<<<dim3(T_DIM / 32, C_DIM / 32, BATCH), 256, 0, stream>>>(x, pe, xt);

  // GEMM1: [16384,512] x [512,1536] -> scatter Q/Kt/V
  ga_gemm_kernel<0><<<dim3(3 * C_DIM / 64, M / 128), 256, 0, stream>>>(
      xt, W1h, b1, Qb, Ktb, Vb, nullptr, M, 3 * C_DIM, C_DIM);

  // flash attention -> overwrites xt region with O[(b,t), c]
  ga_attn_kernel<<<dim3(T_DIM / 64, BATCH * HEADS), 128, 0, stream>>>(Qb, Ktb, Vb, xt);

  // GEMM2: [16384,512] x [512,1024] -> g
  ga_gemm_kernel<1><<<dim3(2 * C_DIM / 64, M / 128), 256, 0, stream>>>(
      xt, W2h, b2, nullptr, nullptr, nullptr, g, M, 2 * C_DIM, C_DIM);

  // gating epilogue
  ga_final_kernel<<<(size_t)M * C_DIM / 256, 256, 0, stream>>>(x, g, out);
}
